// SPACY_GraphAttentionLayer_5935644803255
// MI455X (gfx1250) — matrix operand order repaired and checked by reading, not compiled, not run on hardware
//
#include <hip/hip_runtime.h>

// ---------------------------------------------------------------------------
// GAT layer for MI455X (gfx1250, wave32, WMMA, async-to-LDS).
//   B=8, N=2048, IN_F=256, OUT_F=128.
//   Kernel 0: W1/W2 -> bf16 transposed [col][k] in ws (L2-resident B-matrix).
//   Kernel 1: Wh=h@W1, Wj=j@W2 (bf16 WMMA, f32 acc) -> V=Wh+Wj (bf16 ws),
//             s1=Wh@a1, s2=Wh@a2 (cross-lane reduce). No LDS, no barriers.
//   Kernel 2: flash-style masked softmax over k (adj stream = 134MB = the
//             HBM bound) fused with P@V via v_wmma_f32_16x16x32_bf16.
//             V tiles double-pumped: global_load_async_to_lds_b128 prefetches
//             the next 32xk tile into an LDS bounce buffer while the current
//             tile's WMMA chain runs (ASYNCcnt-tracked, s_wait_asynccnt).
// ---------------------------------------------------------------------------

typedef __attribute__((ext_vector_type(16))) __bf16 v16bf;
typedef __attribute__((ext_vector_type(8)))  __bf16 v8bf;
typedef __attribute__((ext_vector_type(8)))  float  v8f;

#define IN_F   256
#define OUT_F  128
#define NB     8
#define NN     2048
#define LALPHA 0.2f
#define NEG_INF (-9.0e15f)

union AFrag { v16bf v; __bf16 e[16]; };

static __device__ __forceinline__ v16bf concat8(v8bf lo, v8bf hi) {
  return __builtin_shufflevector(lo, hi, 0,1,2,3,4,5,6,7,8,9,10,11,12,13,14,15);
}

// ---------------------------------------------------------------------------
// Kernel 0: pre-convert W1,W2 (f32 [k][col]) -> Wt (bf16 [mat][col][k]).
// 128 KB total, stays in L2 for kernel 1's B-fragment reads.
// ---------------------------------------------------------------------------
__global__ __launch_bounds__(256) void gat_wconv(
    const float* __restrict__ W1, const float* __restrict__ W2,
    __bf16* __restrict__ Wt)
{
  int idx = blockIdx.x * 256 + threadIdx.x;     // 2*128*256 = 65536 launches
  int mat = idx >> 15;
  int rem = idx & 32767;
  int col = rem >> 8;
  int k   = rem & 255;
  const float* W = mat ? W2 : W1;
  Wt[idx] = (__bf16)W[(size_t)k * OUT_F + col];
}

// ---------------------------------------------------------------------------
// Kernel 1: projections. 256 threads = 8 waves; wave w owns rows
// [blk*128 + w*16, +16). B-fragments read straight from transposed bf16 W in
// L2 (two contiguous 16B loads per lane), batched ahead of the WMMA chain.
// ---------------------------------------------------------------------------
__global__ __launch_bounds__(256) void gat_project(
    const float* __restrict__ h, const float* __restrict__ jm,
    const __bf16* __restrict__ Wt,
    const float* __restrict__ av,
    __bf16* __restrict__ V, float* __restrict__ s1, float* __restrict__ s2)
{
  const int tid  = threadIdx.x;
  const int lane = tid & 31;
  const int wave = tid >> 5;
  const int ln   = lane & 15;
  const int half = (lane & 16) >> 1;            // 0 or 8 (row offset, upper lanes)
  const int rowTile = blockIdx.x * 128 + wave * 16;
  const int myRow   = rowTile + ln;             // A-fragment row for this lane
  const int base    = (lane < 16) ? 0 : 8;      // A-frag K base within 32-chunk
  const int kb      = (lane < 16) ? 0 : 8;      // B-frag K base within 32-chunk

  v8f accH[8] = {};
  v8f accJ[8] = {};

  const float*  hrow = h  + (size_t)myRow * IN_F;
  const float*  jrow = jm + (size_t)myRow * IN_F;
  const __bf16* w1l  = Wt + (size_t)ln * IN_F + kb;            // + nt*16*IN_F + kt*32
  const __bf16* w2l  = w1l + (size_t)OUT_F * IN_F;

  for (int kt = 0; kt < 8; ++kt) {
    if (kt + 2 < 8) {
      __builtin_prefetch(hrow + (kt + 2) * 32, 0, 1);
      __builtin_prefetch(jrow + (kt + 2) * 32, 0, 1);
    }

    // ---- A fragments (16x32 bf16): elems 0..7 = K base..base+7,
    //      elems 8..15 = K base+16..base+23 ----
    AFrag aH, aJ;
    {
      float4 f0 = *(const float4*)(hrow + kt * 32 + base);
      float4 f1 = *(const float4*)(hrow + kt * 32 + base + 4);
      float4 f2 = *(const float4*)(hrow + kt * 32 + base + 16);
      float4 f3 = *(const float4*)(hrow + kt * 32 + base + 20);
      aH.e[0]=(__bf16)f0.x; aH.e[1]=(__bf16)f0.y; aH.e[2]=(__bf16)f0.z; aH.e[3]=(__bf16)f0.w;
      aH.e[4]=(__bf16)f1.x; aH.e[5]=(__bf16)f1.y; aH.e[6]=(__bf16)f1.z; aH.e[7]=(__bf16)f1.w;
      aH.e[8]=(__bf16)f2.x; aH.e[9]=(__bf16)f2.y; aH.e[10]=(__bf16)f2.z; aH.e[11]=(__bf16)f2.w;
      aH.e[12]=(__bf16)f3.x; aH.e[13]=(__bf16)f3.y; aH.e[14]=(__bf16)f3.z; aH.e[15]=(__bf16)f3.w;
      f0 = *(const float4*)(jrow + kt * 32 + base);
      f1 = *(const float4*)(jrow + kt * 32 + base + 4);
      f2 = *(const float4*)(jrow + kt * 32 + base + 16);
      f3 = *(const float4*)(jrow + kt * 32 + base + 20);
      aJ.e[0]=(__bf16)f0.x; aJ.e[1]=(__bf16)f0.y; aJ.e[2]=(__bf16)f0.z; aJ.e[3]=(__bf16)f0.w;
      aJ.e[4]=(__bf16)f1.x; aJ.e[5]=(__bf16)f1.y; aJ.e[6]=(__bf16)f1.z; aJ.e[7]=(__bf16)f1.w;
      aJ.e[8]=(__bf16)f2.x; aJ.e[9]=(__bf16)f2.y; aJ.e[10]=(__bf16)f2.z; aJ.e[11]=(__bf16)f2.w;
      aJ.e[12]=(__bf16)f3.x; aJ.e[13]=(__bf16)f3.y; aJ.e[14]=(__bf16)f3.z; aJ.e[15]=(__bf16)f3.w;
    }

    // ---- batch all 8 B-fragments (W1), then the WMMA chain ----
    v16bf Bf[8];
    #pragma unroll
    for (int nt = 0; nt < 8; ++nt) {
      const __bf16* p = w1l + (size_t)nt * 16 * IN_F + kt * 32;
      Bf[nt] = concat8(*(const v8bf*)p, *(const v8bf*)(p + 16));
    }
    #pragma unroll
    for (int nt = 0; nt < 8; ++nt)
      accH[nt] = __builtin_amdgcn_wmma_f32_16x16x32_bf16(
          false, aH.v, false, Bf[nt], (short)0, accH[nt], false, false);

    #pragma unroll
    for (int nt = 0; nt < 8; ++nt) {
      const __bf16* p = w2l + (size_t)nt * 16 * IN_F + kt * 32;
      Bf[nt] = concat8(*(const v8bf*)p, *(const v8bf*)(p + 16));
    }
    #pragma unroll
    for (int nt = 0; nt < 8; ++nt)
      accJ[nt] = __builtin_amdgcn_wmma_f32_16x16x32_bf16(
          false, aJ.v, false, Bf[nt], (short)0, accJ[nt], false, false);
  }

  // ---- s1 = Wh@a1, s2 = Wh@a2 (reduce over columns = across 16 lanes) ----
  float a1v[8], a2v[8];
  #pragma unroll
  for (int nt = 0; nt < 8; ++nt) {
    a1v[nt] = av[nt * 16 + ln];
    a2v[nt] = av[OUT_F + nt * 16 + ln];
  }

  #pragma unroll
  for (int r = 0; r < 8; ++r) {
    float p1 = 0.f, p2 = 0.f;
    #pragma unroll
    for (int nt = 0; nt < 8; ++nt) {
      p1 += accH[nt][r] * a1v[nt];
      p2 += accH[nt][r] * a2v[nt];
    }
    #pragma unroll
    for (int msk = 1; msk <= 8; msk <<= 1) {
      p1 += __shfl_xor(p1, msk, 32);
      p2 += __shfl_xor(p2, msk, 32);
    }
    const int rowg = rowTile + r + half;         // C layout: VGPR r -> row r / r+8
    if (ln == 0) { s1[rowg] = p1; s2[rowg] = p2; }
    #pragma unroll
    for (int nt = 0; nt < 8; ++nt)
      V[(size_t)rowg * OUT_F + nt * 16 + ln] = (__bf16)(accH[nt][r] + accJ[nt][r]);
  }
}

// ---------------------------------------------------------------------------
// Kernel 2: masked softmax(row) @ V, fused & streamed (flash style).
// 128 threads = 4 waves; wave owns 16 rows; block owns 64 rows of one batch.
// V tile for kt+1 is async-copied into LDS (Vraw, row-major) while kt's WMMA
// chain runs; each iteration starts by transposing Vraw -> Vt (B-frag layout).
// ---------------------------------------------------------------------------
__global__ __launch_bounds__(128) void gat_attention(
    const int* __restrict__ adj, const __bf16* __restrict__ V,
    const float* __restrict__ s1, const float* __restrict__ s2,
    float* __restrict__ out)
{
  __shared__ __bf16 Vraw[32][128];               // async landing zone (row-major)
  __shared__ __bf16 Vt[128][40];                 // [col][k], padded, B-frag layout
  __shared__ float  ss2[32];

  const int tid  = threadIdx.x;
  const int lane = tid & 31;
  const int wave = tid >> 5;
  const int ln   = lane & 15;
  const int half = (lane & 16) >> 1;             // 0 or 8
  const int b     = blockIdx.x >> 5;
  const int strip = blockIdx.x & 31;
  const int rowTile = strip * 64 + wave * 16;
  const int myRow   = rowTile + ln;
  const int base    = (lane < 16) ? 0 : 8;       // A-frag K base
  const int kb      = (lane < 16) ? 0 : 8;       // B-frag K base

  const float s1row = s1[(size_t)b * NN + myRow];
  const int* adjRow = adj + ((size_t)b * NN + myRow) * NN;
  const __bf16* Vb  = V + (size_t)b * NN * OUT_F;

  // per-thread async-copy slots: 4 x 16B chunks of the 32x128 bf16 tile
  const unsigned vrawBase = (unsigned)(size_t)(&Vraw[0][0]);

  #define STAGE_ASYNC(K0)                                                      \
    do {                                                                       \
      _Pragma("unroll")                                                        \
      for (int it = 0; it < 4; ++it) {                                         \
        int c    = tid + it * 128;               /* 512 chunks of 16B */       \
        int kloc = c >> 4;                                                     \
        int celt = (c & 15) * 8;                                               \
        const __bf16* gp = Vb + ((size_t)((K0) + kloc)) * OUT_F + celt;        \
        unsigned loff = vrawBase + (unsigned)c * 16u;                          \
        asm volatile("global_load_async_to_lds_b128 %0, %1, off"               \
                     :: "v"(loff), "v"(gp) : "memory");                        \
      }                                                                        \
    } while (0)

  float m = -1.0e30f, lsum = 0.f;
  v8f acc[8] = {};

  // prologue: pull tile 0
  STAGE_ASYNC(0);
  asm volatile("s_wait_asynccnt 0x0" ::: "memory");

  for (int kt = 0; kt < 64; ++kt) {
    const int k0 = kt * 32;

    __syncthreads();                             // Vraw(kt) complete, Vt free

    // ---- transpose Vraw -> Vt; stage s2 tile ----
    if (tid < 32) ss2[tid] = s2[(size_t)b * NN + k0 + tid];
    #pragma unroll
    for (int it = 0; it < 4; ++it) {
      int c    = tid + it * 128;
      int kloc = c >> 4;
      int celt = (c & 15) * 8;
      v8bf vv = *(const v8bf*)&Vraw[kloc][celt];
      #pragma unroll
      for (int i = 0; i < 8; ++i) Vt[celt + i][kloc] = vv[i];
    }
    __syncthreads();                             // Vt ready, Vraw reusable

    if (kt + 1 < 64) STAGE_ASYNC(k0 + 32);       // overlap next tile with compute

    if (kt + 8 < 64)                             // stream adj well ahead
      __builtin_prefetch(adjRow + k0 + 8 * 32, 0, 1);

    // ---- masked logits for this lane's 16 K values (adj = the HBM stream;
    //      lane pair l/l+16 covers this row's 128 contiguous bytes) ----
    int4 ad0 = *(const int4*)(adjRow + k0 + base);
    int4 ad1 = *(const int4*)(adjRow + k0 + base + 4);
    int4 ad2 = *(const int4*)(adjRow + k0 + base + 16);
    int4 ad3 = *(const int4*)(adjRow + k0 + base + 20);
    int avj[16] = { ad0.x, ad0.y, ad0.z, ad0.w, ad1.x, ad1.y, ad1.z, ad1.w,
                    ad2.x, ad2.y, ad2.z, ad2.w, ad3.x, ad3.y, ad3.z, ad3.w };
    float ev[16];
    float tmax = -1.0e30f;
    #pragma unroll
    for (int i = 0; i < 8; ++i) {
      float sv = s1row + ss2[base + i];
      float e  = (sv > 0.f) ? sv : LALPHA * sv;
      ev[i] = (avj[i] > 0) ? e : NEG_INF;
      float sv2 = s1row + ss2[base + 16 + i];
      float e2  = (sv2 > 0.f) ? sv2 : LALPHA * sv2;
      ev[8 + i] = (avj[8 + i] > 0) ? e2 : NEG_INF;
    }
    #pragma unroll
    for (int i = 0; i < 16; ++i) tmax = fmaxf(tmax, ev[i]);
    tmax = fmaxf(tmax, __shfl_xor(tmax, 16, 32));

    const float mnew  = fmaxf(m, tmax);
    const float alpha = __expf(m - mnew);

    AFrag aP;
    float psum = 0.f;
    #pragma unroll
    for (int i = 0; i < 16; ++i) {
      float p = __expf(ev[i] - mnew);
      psum += p;
      aP.e[i] = (__bf16)p;
    }
    psum += __shfl_xor(psum, 16, 32);
    lsum = lsum * alpha + psum;
    m = mnew;

    // row-broadcast of alpha once per r (C layout: VGPR r -> row r / r+8)
    float asc[8];
    #pragma unroll
    for (int r = 0; r < 8; ++r) asc[r] = __shfl(alpha, r + half, 32);

    // ---- batch B-fragments, then rescale + WMMA chain ----
    v16bf bv[8];
    #pragma unroll
    for (int nt = 0; nt < 8; ++nt) {
      int col = nt * 16 + ln;
      bv[nt] = concat8(*(const v8bf*)&Vt[col][kb], *(const v8bf*)&Vt[col][kb + 16]);
    }
    #pragma unroll
    for (int nt = 0; nt < 8; ++nt) {
      #pragma unroll
      for (int r = 0; r < 8; ++r) acc[nt][r] *= asc[r];
      acc[nt] = __builtin_amdgcn_wmma_f32_16x16x32_bf16(
          false, aP.v, false, bv[nt], (short)0, acc[nt], false, false);
    }

    // my async for kt+1 must be done before next top-of-loop barrier
    asm volatile("s_wait_asynccnt 0x0" ::: "memory");
  }
  #undef STAGE_ASYNC

  // ---- normalize, ELU, store ----
  const float inv = 1.0f / lsum;
  float isc[8];
  #pragma unroll
  for (int r = 0; r < 8; ++r) isc[r] = __shfl(inv, r + half, 32);
  #pragma unroll
  for (int nt = 0; nt < 8; ++nt) {
    #pragma unroll
    for (int r = 0; r < 8; ++r) {
      float val = acc[nt][r] * isc[r];
      val = (val > 0.f) ? val : (__expf(val) - 1.0f);
      out[((size_t)b * NN + rowTile + r + half) * OUT_F + nt * 16 + ln] = val;
    }
  }
}

// ---------------------------------------------------------------------------
extern "C" void kernel_launch(void* const* d_in, const int* in_sizes, int n_in,
                              void* d_out, int out_size, void* d_ws, size_t ws_size,
                              hipStream_t stream) {
  const float* h   = (const float*)d_in[0];
  const float* jm  = (const float*)d_in[1];
  const int*   adj = (const int*)d_in[2];
  const float* W1  = (const float*)d_in[3];
  const float* W2  = (const float*)d_in[4];
  const float* av  = (const float*)d_in[5];
  float* out = (float*)d_out;

  char* ws = (char*)d_ws;
  __bf16* V  = (__bf16*)ws;                                   // 4 MB
  float*  s1 = (float*)(ws + (size_t)NB * NN * OUT_F * 2);    // 64 KB
  float*  s2 = s1 + NB * NN;                                  // 64 KB
  __bf16* Wt = (__bf16*)(s2 + NB * NN);                       // 128 KB

  gat_wconv<<<dim3(256), dim3(256), 0, stream>>>(W1, W2, Wt);
  gat_project<<<dim3((NB * NN) / 128), dim3(256), 0, stream>>>(
      h, jm, Wt, av, V, s1, s2);
  gat_attention<<<dim3(NB * (NN / 64)), dim3(128), 0, stream>>>(
      adj, V, s1, s2, out);
}
